// Model_59622736003377
// MI455X (gfx1250) — matrix operand order repaired and checked by reading, not compiled, not run on hardware
//
#include <hip/hip_runtime.h>

// ---------------------------------------------------------------------------
// Problem constants (from reference): V=4, N=1536, D=256, HEADS=4, topk=32
// ---------------------------------------------------------------------------
#define VV 4
#define NN 1536
#define DD 256
#define NH 4
#define DH 64
#define TOPK 32

typedef __attribute__((ext_vector_type(16))) __bf16 bf16x16;
typedef __attribute__((ext_vector_type(8)))  __bf16 bf16x8;
typedef __attribute__((ext_vector_type(8)))  float  f32x8;

__device__ __forceinline__ __bf16 f2bf(float f) {
    union { float f; unsigned u; } x; x.f = f;
    unsigned r = x.u + 0x7FFFu + ((x.u >> 16) & 1u);   // round-to-nearest-even
    unsigned short h = (unsigned short)(r >> 16);
    __bf16 b;
    __builtin_memcpy(&b, &h, 2);
    return b;
}

// ---------------------------------------------------------------------------
// WMMA GEMM:  Out[M x Nout] = A[M x KDIM] * B[Nout x KDIM]^T  (+ epilogue)
//   MODE 0: Out = acc + bias[col]                              (qkv proj)
//   MODE 1: a = alphas[row/N]; Out = a*(acc+bias) + (1-a)*Hres (aligned)
//   MODE 2: Out = acc        (Qn/Kn/Vn, batched over blockIdx.z)
// Tile: 128(M) x 64(N), K-step 32, KDIM compile-time -> fully unrolled,
// double-buffered LDS so next tile's global loads overlap the WMMAs.
// 8 wave32 per block; each wave owns a 16x64 strip = 4 WMMA per K-step.
// ---------------------------------------------------------------------------
template <int MODE, int KDIM>
__global__ __launch_bounds__(256) void gemm_wmma(
    const float* __restrict__ A, const float* __restrict__ B,
    const float* __restrict__ bias, float* __restrict__ Out,
    int M, int Nout,
    const float* __restrict__ Hres, const float* __restrict__ alphas,
    long bsA, long bsB, long bsO)
{
    A   += (long)blockIdx.z * bsA;
    B   += (long)blockIdx.z * bsB;
    Out += (long)blockIdx.z * bsO;

    __shared__ alignas(16) __bf16 As[2][128][32];   // 2 x 8 KB
    __shared__ alignas(16) __bf16 Bs[2][64][32];    // 2 x 4 KB

    const int tid  = threadIdx.x;
    const int lane = tid & 31;
    const int wave = tid >> 5;                   // 0..7
    const int m0   = blockIdx.x * 128;
    const int n0   = blockIdx.y * 64;

    constexpr int KSTEPS = KDIM / 32;

    auto load_tiles = [&](int k0, int buf) {
        // A tile (128x32 f32 -> bf16): 1024 float4, 4 per thread
        #pragma unroll
        for (int it = 0; it < 4; ++it) {
            int id = tid + it * 256;             // float4 index
            int r  = id >> 3, c4 = (id & 7) << 2;
            float4 v = *(const float4*)(A + (long)(m0 + r) * KDIM + k0 + c4);
            As[buf][r][c4 + 0] = f2bf(v.x); As[buf][r][c4 + 1] = f2bf(v.y);
            As[buf][r][c4 + 2] = f2bf(v.z); As[buf][r][c4 + 3] = f2bf(v.w);
        }
        // B tile (64x32 f32 -> bf16): 512 float4, 2 per thread
        #pragma unroll
        for (int it = 0; it < 2; ++it) {
            int id = tid + it * 256;
            int r  = id >> 3, c4 = (id & 7) << 2;
            float4 v = *(const float4*)(B + (long)(n0 + r) * KDIM + k0 + c4);
            Bs[buf][r][c4 + 0] = f2bf(v.x); Bs[buf][r][c4 + 1] = f2bf(v.y);
            Bs[buf][r][c4 + 2] = f2bf(v.z); Bs[buf][r][c4 + 3] = f2bf(v.w);
        }
    };

    f32x8 acc[4] = {};

    load_tiles(0, 0);
    __syncthreads();

    #pragma unroll
    for (int s = 0; s < KSTEPS; ++s) {
        const int buf = s & 1;
        if (s + 1 < KSTEPS) load_tiles((s + 1) * 32, buf ^ 1);

        // --- A fragment (ISA 7.12.2, 16-bit A 16x32): lane = M,
        //     K elements: {kb..kb+7, kb+16..kb+23}, kb = (lane&16)?8:0
        int am = wave * 16 + (lane & 15);
        int kb = (lane & 16) ? 8 : 0;
        bf16x8 alo = *(const bf16x8*)&As[buf][am][kb];
        bf16x8 ahi = *(const bf16x8*)&As[buf][am][kb + 16];
        bf16x16 afrag = __builtin_shufflevector(alo, ahi,
            0,1,2,3,4,5,6,7,8,9,10,11,12,13,14,15);

        #pragma unroll
        for (int t = 0; t < 4; ++t) {
            // --- B fragment (32x16): lane = N, K elements: {kb..kb+7, kb+16..kb+23}, kb = (lane&16)?8:0
            int bn = t * 16 + (lane & 15);
            int bk = (lane & 16) ? 8 : 0;
            bf16x8 blo = *(const bf16x8*)&Bs[buf][bn][bk];
            bf16x8 bhi = *(const bf16x8*)&Bs[buf][bn][bk + 16];
            bf16x16 bfrag = __builtin_shufflevector(blo, bhi,
                0,1,2,3,4,5,6,7,8,9,10,11,12,13,14,15);
            acc[t] = __builtin_amdgcn_wmma_f32_16x16x32_bf16(
                false, afrag, false, bfrag, (short)0, acc[t], false, false);
        }
        __syncthreads();   // protects buf (to be overwritten) and buf^1 fill
    }

    // --- epilogue: D layout = VGPR r -> M = r + (lane&16?8:0), N = lane&15
    int mbase = m0 + wave * 16 + ((lane & 16) ? 8 : 0);
    int nbase = n0 + (lane & 15);
    #pragma unroll
    for (int t = 0; t < 4; ++t) {
        int col = nbase + t * 16;
        float bcol = (MODE == 2) ? 0.0f : bias[col];
        #pragma unroll
        for (int r = 0; r < 8; ++r) {
            int row = mbase + r;
            float val = acc[t][r] + bcol;
            if (MODE == 1) {
                float a = alphas[row / NN];
                val = a * val + (1.0f - a) * Hres[(long)row * Nout + col];
            }
            Out[(long)row * Nout + col] = val;
        }
    }
}

// ---------------------------------------------------------------------------
// Intra-view 4x4 attention per (n, head): o = softmax(q k^T/sqrt(64)) v
// ---------------------------------------------------------------------------
__global__ __launch_bounds__(64) void attn_intra(
    const float* __restrict__ qkv, float* __restrict__ o)
{
    int n = blockIdx.x >> 2;
    int h = blockIdx.x & 3;
    int t = threadIdx.x;                 // d within head, 0..63
    __shared__ float q[VV][DH], k[VV][DH], v[VV][DH], w[VV][VV];

    #pragma unroll
    for (int s = 0; s < VV; ++s) {
        long base = ((long)(s * NN + n)) * (3 * DD) + h * DH + t;
        q[s][t] = qkv[base];
        k[s][t] = qkv[base + DD];
        v[s][t] = qkv[base + 2 * DD];
    }
    __syncthreads();
    if (t < 16) {
        int s = t >> 2, tt = t & 3;
        float acc = 0.0f;
        for (int d = 0; d < DH; ++d) acc += q[s][d] * k[tt][d];
        w[s][tt] = acc * 0.125f;         // 1/sqrt(64)
    }
    __syncthreads();
    if (t < 4) {
        float mx = fmaxf(fmaxf(w[t][0], w[t][1]), fmaxf(w[t][2], w[t][3]));
        float e0 = __expf(w[t][0] - mx), e1 = __expf(w[t][1] - mx);
        float e2 = __expf(w[t][2] - mx), e3 = __expf(w[t][3] - mx);
        float inv = 1.0f / (e0 + e1 + e2 + e3);
        w[t][0] = e0 * inv; w[t][1] = e1 * inv; w[t][2] = e2 * inv; w[t][3] = e3 * inv;
    }
    __syncthreads();
    #pragma unroll
    for (int s = 0; s < VV; ++s) {
        float acc = w[s][0] * v[0][t] + w[s][1] * v[1][t]
                  + w[s][2] * v[2][t] + w[s][3] * v[3][t];
        o[((long)(s * NN + n)) * DD + h * DH + t] = acc;
    }
}

// ---------------------------------------------------------------------------
// Top-32 of each length-1536 row of C; write normalized values + indices.
// One 256-thread block per row. The row is staged into LDS with gfx1250
// async copies (global_load_async_to_lds_b128, ASYNCcnt-tracked): each lane
// moves 16 B memory->LDS with no VGPR round trip; then 32 argmax reductions.
// ---------------------------------------------------------------------------
__global__ __launch_bounds__(256) void topk_kernel(
    const float* __restrict__ C, float* __restrict__ tv, int* __restrict__ ti)
{
    long row = blockIdx.x;
    const float* crow = C + row * NN;
    __shared__ alignas(16) float vals[NN];
    __shared__ float rv[256];
    __shared__ int   ri[256];
    __shared__ float selv[TOPK];
    __shared__ int   seli[TOPK];
    __shared__ float ssum;
    int t = threadIdx.x;

    // async stage: 384 float4 (6 KB) memory -> LDS
    for (int i = t; i < NN / 4; i += 256) {
        unsigned lds = (unsigned)(size_t)&vals[i * 4];            // LDS byte offset
        unsigned long long ga = (unsigned long long)(size_t)(crow + i * 4);
        asm volatile("global_load_async_to_lds_b128 %0, %1, off"
                     :: "v"(lds), "v"(ga) : "memory");
    }
    asm volatile("s_wait_asynccnt 0x0" ::: "memory");
    __syncthreads();

    for (int it = 0; it < TOPK; ++it) {
        float best = -3.4e38f; int bi = 0;
        for (int i = t; i < NN; i += 256) {
            float x = vals[i];
            if (x > best) { best = x; bi = i; }
        }
        rv[t] = best; ri[t] = bi;
        __syncthreads();
        for (int s = 128; s > 0; s >>= 1) {
            if (t < s && rv[t + s] > rv[t]) { rv[t] = rv[t + s]; ri[t] = ri[t + s]; }
            __syncthreads();
        }
        if (t == 0) { selv[it] = rv[0]; seli[it] = ri[0]; vals[ri[0]] = -3.4e38f; }
        __syncthreads();
    }
    if (t == 0) {
        float s = 0.0f;
        for (int i = 0; i < TOPK; ++i) s += selv[i];
        ssum = s + 1e-12f;
    }
    __syncthreads();
    if (t < TOPK) {
        tv[row * TOPK + t] = selv[t] / ssum;   // only the sign (mask) matters
        ti[row * TOPK + t] = seli[t];
    }
}

// ---------------------------------------------------------------------------
// Column-mean of Vn per view (for the fully-masked p==q uniform-softmax term)
// ---------------------------------------------------------------------------
__global__ __launch_bounds__(256) void vmean_kernel(
    const float* __restrict__ Vn, float* __restrict__ Vm)
{
    int p = blockIdx.x, t = threadIdx.x;
    const float* base = Vn + (long)p * NN * DD + t;
    float s = 0.0f;
    for (int m = 0; m < NN; ++m) s += base[(long)m * DD];
    Vm[p * DD + t] = s * (1.0f / (float)NN);
}

// ---------------------------------------------------------------------------
// Sparse cross-view attention + final fuse. One block per (p, n):
//  - gather 3*32 top-k scores vs Kn (dense score GEMM is skipped entirely),
//  - per-(p,q) softmax over 32 entries, weighted gather of Vn,
//  - add uniform p==q term (Vmean), relu-blend, write d_out.
// ---------------------------------------------------------------------------
__global__ __launch_bounds__(256) void sparse_attn(
    const float* __restrict__ Qn, const float* __restrict__ Kn,
    const float* __restrict__ Vn, const float* __restrict__ Vm,
    const float* __restrict__ tv, const int* __restrict__ ti,
    const float* __restrict__ aligned, const float* __restrict__ Hin,
    const float* __restrict__ alpha_align, const float* __restrict__ beta,
    float* __restrict__ out)
{
    int p = blockIdx.x / NN;
    int n = blockIdx.x - p * NN;
    int t = threadIdx.x;

    __shared__ alignas(16) float qv[DD];
    __shared__ float sc[96];
    __shared__ float wts[96];
    __shared__ long  off[96];

    qv[t] = Qn[((long)(p * NN + n)) * DD + t];
    __syncthreads();

    if (t < 96) {
        int gi = t >> 5;                       // 0..2 -> the 3 views q != p
        int j  = t & 31;
        int q  = gi + (gi >= p ? 1 : 0);
        long trow = ((long)(p * VV + q) * NN + n);
        float val = tv[trow * TOPK + j];
        int   m   = ti[trow * TOPK + j];
        long  ko  = ((long)(q * NN + m)) * DD;
        float s;
        if (val > 0.0f) {                      // mask = (C_hat > 0)
            const float4* k4 = (const float4*)(Kn + ko);
            const float4* q4 = (const float4*)qv;
            s = 0.0f;
            for (int d = 0; d < DD / 4; ++d) {
                float4 a = q4[d], b = k4[d];
                s += a.x * b.x + a.y * b.y + a.z * b.z + a.w * b.w;
            }
            s *= 0.0625f;                      // 1/sqrt(256)
        } else {
            s = -1e30f;                        // masked -> exp underflows to 0
        }
        sc[t] = s; off[t] = ko;
    }
    __syncthreads();
    if (t < 3) {                               // per-(p,q) softmax over 32
        int b = t * 32;
        float mx = -1e30f;
        for (int j = 0; j < 32; ++j) mx = fmaxf(mx, sc[b + j]);
        float sum = 0.0f;
        for (int j = 0; j < 32; ++j) { float e = __expf(sc[b + j] - mx); wts[b + j] = e; sum += e; }
        float inv = (sum > 0.0f) ? 1.0f / sum : 0.0f;
        for (int j = 0; j < 32; ++j) wts[b + j] *= inv;
    }
    __syncthreads();

    float acc = Vm[p * DD + t];                // p==q uniform-softmax term
    for (int g = 0; g < 96; ++g) acc += wts[g] * Vn[off[g] + t];

    long i = ((long)(p * NN + n)) * DD + t;
    float aa = 1.0f / (1.0f + __expf(-alpha_align[0]));
    float f  = aa * aligned[i] + (1.0f - aa) * acc;
    f = fmaxf(f, 0.0f);
    float bt = beta[0];
    out[i] = bt * Hin[i] + (1.0f - bt) * f;
}

// ---------------------------------------------------------------------------
// Workspace layout (floats). Regions reused once their producer is consumed:
//   [0, 4718592)        qkv (6144x768)      -> later Qn/Kn/Vn
//   [4718592, 6291456)  o   (6144x256)      -> later topk vals + indices
//   [6291456, 7864320)  aligned (6144x256)
//   [7864320, 7865344)  Vmean (4x256)
// Total ~30 MB.
// ---------------------------------------------------------------------------
extern "C" void kernel_launch(void* const* d_in, const int* in_sizes, int n_in,
                              void* d_out, int out_size, void* d_ws, size_t ws_size,
                              hipStream_t stream) {
    (void)in_sizes; (void)n_in; (void)out_size; (void)ws_size;
    const float* H      = (const float*)d_in[0];
    const float* C      = (const float*)d_in[1];
    const float* WQ     = (const float*)d_in[2];
    const float* WK     = (const float*)d_in[3];
    const float* WVp    = (const float*)d_in[4];
    const float* ipw    = (const float*)d_in[5];
    const float* ipb    = (const float*)d_in[6];
    const float* opw    = (const float*)d_in[7];
    const float* opb    = (const float*)d_in[8];
    const float* alphas = (const float*)d_in[9];
    const float* aalign = (const float*)d_in[10];
    const float* beta   = (const float*)d_in[11];

    float* ws      = (float*)d_ws;
    float* qkv     = ws;                       // 4,718,592 floats
    float* o_      = ws + 4718592;             // 1,572,864 floats
    float* aligned = ws + 6291456;             // 1,572,864 floats
    float* vmean   = ws + 7864320;             // 1,024 floats
    float* Qn      = ws;                       // reuse of qkv region
    float* Kn      = ws + 1572864;
    float* Vn      = ws + 3145728;
    float* tv      = o_;                       // reuse of o region
    int*   ti      = (int*)(o_ + 786432);

    dim3 b256(256);
    const long NB = (long)NN * DD;             // per-view batch stride
    const long WB = (long)DD * DD;

    // 1) qkv = H(6144x256) * in_proj_w^T + b        (wmma, MODE 0)
    gemm_wmma<0, DD><<<dim3(48, 12, 1), b256, 0, stream>>>(
        H, ipw, ipb, qkv, VV * NN, 3 * DD, nullptr, nullptr, 0, 0, 0);

    // 2) tiny 4x4 intra-view attention -> o
    attn_intra<<<dim3(NN * NH), dim3(64), 0, stream>>>(qkv, o_);

    // 3) aligned = a*(o * out_proj^T + b) + (1-a)*H (wmma, MODE 1)
    gemm_wmma<1, DD><<<dim3(48, 4, 1), b256, 0, stream>>>(
        o_, opw, opb, aligned, VV * NN, DD, H, alphas, 0, 0, 0);

    // 4) Qn/Kn/Vn = per-view aligned * W^T          (wmma, MODE 2, z = view)
    gemm_wmma<2, DD><<<dim3(12, 4, VV), b256, 0, stream>>>(
        aligned, WQ, nullptr, Qn, NN, DD, nullptr, nullptr, NB, WB, NB);
    gemm_wmma<2, DD><<<dim3(12, 4, VV), b256, 0, stream>>>(
        aligned, WK, nullptr, Kn, NN, DD, nullptr, nullptr, NB, WB, NB);
    gemm_wmma<2, DD><<<dim3(12, 4, VV), b256, 0, stream>>>(
        aligned, WVp, nullptr, Vn, NN, DD, nullptr, nullptr, NB, WB, NB);

    // 5) top-32 per row of C (dominant HBM term: 151 MB read once, async->LDS)
    topk_kernel<<<dim3(VV * VV * NN), b256, 0, stream>>>(C, tv, ti);

    // 6) column means of Vn (p==q uniform term)
    vmean_kernel<<<dim3(VV), b256, 0, stream>>>(Vn, vmean);

    // 7) sparse attention + final fuse -> d_out
    sparse_attn<<<dim3(VV * NN), b256, 0, stream>>>(
        Qn, Kn, Vn, vmean, tv, ti, aligned, H, aalign, beta, (float*)d_out);
}
